// GPT_32890859553076
// MI455X (gfx1250) — matrix operand order repaired and checked by reading, not compiled, not run on hardware
//
#include <hip/hip_runtime.h>
#include <hip/hip_bf16.h>

// ---------------------------------------------------------------------------
// GPT forward for MI455X (gfx1250, wave32, WMMA).
// All GEMMs run as bf16 x bf16 -> f32 via V_WMMA_F32_16X16X32_BF16.
// A-tile staging uses the Tensor Data Mover (tensor_load_to_lds + TENSORcnt)
// when the builtin is available; attention V goes through LDS transposed.
// ---------------------------------------------------------------------------

typedef __bf16  bf16_t;
typedef __attribute__((ext_vector_type(16))) __bf16 v16bf;
typedef __attribute__((ext_vector_type(8)))  float  v8f;
typedef __attribute__((ext_vector_type(4)))  unsigned int u32x4;
typedef __attribute__((ext_vector_type(8)))  int i32x8;
typedef __attribute__((ext_vector_type(4)))  int i32x4;

#if __has_builtin(__builtin_amdgcn_tensor_load_to_lds) && \
    __has_builtin(__builtin_amdgcn_s_wait_tensorcnt)
#define USE_TDM 1
#else
#define USE_TDM 0
#endif

#define GPT_L 4
#define GPT_B 4
#define GPT_T 1024
#define GPT_D 1024
#define GPT_H 16
#define GPT_V 32000
#define GPT_DH 64
#define GPT_M (GPT_B * GPT_T)   // 4096 token rows

// ---------------- bf16 helpers (RNE convert, no bf16 arithmetic) ------------
__device__ __forceinline__ unsigned short f2bfbits(float f) {
  unsigned u = __builtin_bit_cast(unsigned, f);
  unsigned r = u + 0x7fffu + ((u >> 16) & 1u);
  return (unsigned short)(r >> 16);
}
__device__ __forceinline__ bf16_t f2bf(float f) {
  unsigned short h = f2bfbits(f);
  return __builtin_bit_cast(bf16_t, h);
}
__device__ __forceinline__ void store4bf(bf16_t* p, float a, float b, float c, float d) {
  unsigned long long pk =
      (unsigned long long)f2bfbits(a) |
      ((unsigned long long)f2bfbits(b) << 16) |
      ((unsigned long long)f2bfbits(c) << 32) |
      ((unsigned long long)f2bfbits(d) << 48);
  *reinterpret_cast<unsigned long long*>(p) = pk;
}

// ---------------- WMMA fragment loaders (ISA 7.12.2 layouts) ----------------
// A 16x32 bf16: lanes 0-15 row M=lane, V0..3 = K{0..7}, V4..7 = K{16..23};
// lanes 16-31 same rows, K offset +8.
__device__ __forceinline__ v16bf load_afrag_row(const bf16_t* pr, int lhalf) {
  v16bf a;
#pragma unroll
  for (int v = 0; v < 4; ++v) {
    int kk = 2 * v + 8 * lhalf;
    a[2 * v] = pr[kk]; a[2 * v + 1] = pr[kk + 1];
  }
#pragma unroll
  for (int v = 4; v < 8; ++v) {
    int kk = 2 * (v - 4) + 16 + 8 * lhalf;
    a[2 * v] = pr[kk]; a[2 * v + 1] = pr[kk + 1];
  }
  return a;
}
// B 32x16 bf16: lane = column N; lanes 0-15 V0..3 = K{0..7}, V4..7 = K{16..23};
// lanes 16-31 K offset +8 (matches A). pc = column data contiguous in K.
__device__ __forceinline__ v16bf load_bfrag_col(const bf16_t* pc, int lhalf) {
  v16bf b;
#pragma unroll
  for (int v = 0; v < 8; ++v) {
    int kk = (v < 4 ? 2 * v : 16 + 2 * (v - 4)) + 8 * lhalf;
    b[2 * v] = pc[kk]; b[2 * v + 1] = pc[kk + 1];
  }
  return b;
}

#if USE_TDM
// Issue a TDM 2D tile load: tile (tileK x tileRows) of bf16 from a row-major
// tensor with row stride strideK (elements), into LDS at byte offset ldsOff.
// Descriptor bitfields per CDNA5 ISA ch.8 (D# group0/group1, 2D, no padding).
// Toolchain here exposes the 6-arg builtin:
//   (u32x4 g0, i32x8 g1, i32x4 g2, i32x4 g3, i32x8 extra, i32 cpol)
__device__ __forceinline__ void tdm_load_2d(unsigned ldsOff,
                                            const bf16_t* gaddr,
                                            unsigned tensorK, unsigned rows,
                                            unsigned tileK, unsigned tileRows,
                                            unsigned long long strideK) {
  unsigned long long ga = (unsigned long long)(size_t)gaddr;
  u32x4 g0 = { 1u,                                    // count=1, user desc
               ldsOff,                                // lds_addr (bytes)
               (unsigned)(ga & 0xffffffffu),
               ((unsigned)(ga >> 32) & 0x01ffffffu) | 0x80000000u }; // type=2
  i32x8 g1 = { (int)(1u << 16),                       // data_size=1 (2 bytes)
               (int)((tensorK & 0xffffu) << 16),      // tensor_dim0 lo
               (int)((tensorK >> 16) | ((rows & 0xffffu) << 16)),
               (int)((rows >> 16) | (tileK << 16)),   // tile_dim0
               (int)(tileRows),                       // tile_dim1 (tile_dim2=0)
               (int)(strideK & 0xffffffffull),        // dim0_stride lo
               (int)((strideK >> 32) & 0xffffull),    // dim0_stride hi
               0 };
  i32x4 z4 = {0, 0, 0, 0};
  i32x8 z8 = {0, 0, 0, 0, 0, 0, 0, 0};
  __builtin_amdgcn_tensor_load_to_lds(g0, g1, z4, z4, z8, 0);
}
#endif

// ---------------- embedding gather: x = wte[idx] ----------------------------
__global__ __launch_bounds__(256) void k_embed(const int* __restrict__ idx,
                                               const float* __restrict__ wte,
                                               float* __restrict__ x) {
  int i = blockIdx.x * 256 + threadIdx.x;          // over M*D/4 float4s
  if (i >= GPT_M * GPT_D / 4) return;
  int bt = i >> 8;                                 // 256 float4 per row
  int c4 = i & 255;
  int row = idx[bt];
  reinterpret_cast<float4*>(x)[i] =
      reinterpret_cast<const float4*>(wte)[(size_t)row * 256 + c4];
}

// ---------------- f32 -> bf16 bulk convert (weight staging) -----------------
__global__ __launch_bounds__(256) void k_f32_to_bf16(const float* __restrict__ in,
                                                     bf16_t* __restrict__ out,
                                                     int n4) {
  int i = blockIdx.x * 256 + threadIdx.x;
  if (i >= n4) return;
  float4 v = reinterpret_cast<const float4*>(in)[i];
  store4bf(out + 4 * (size_t)i, v.x, v.y, v.z, v.w);
}

// ---------------- LayerNorm (row = 1024), out bf16 --------------------------
__global__ __launch_bounds__(256) void k_layernorm(const float* __restrict__ x,
                                                   const float* __restrict__ w,
                                                   const float* __restrict__ b,
                                                   bf16_t* __restrict__ out) {
  __shared__ float r1[256];
  __shared__ float r2[256];
  const int row = blockIdx.x;
  const int tid = threadIdx.x;
  float4 v = reinterpret_cast<const float4*>(x + (size_t)row * GPT_D)[tid];
  float s = v.x + v.y + v.z + v.w;
  float q = v.x * v.x + v.y * v.y + v.z * v.z + v.w * v.w;
  r1[tid] = s; r2[tid] = q;
  __syncthreads();
  for (int o = 128; o > 0; o >>= 1) {
    if (tid < o) { r1[tid] += r1[tid + o]; r2[tid] += r2[tid + o]; }
    __syncthreads();
  }
  float mu = r1[0] * (1.0f / GPT_D);
  float var = r2[0] * (1.0f / GPT_D) - mu * mu;
  float rs = rsqrtf(var + 1e-5f);
  float4 wv = reinterpret_cast<const float4*>(w)[tid];
  float4 bv = reinterpret_cast<const float4*>(b)[tid];
  store4bf(out + (size_t)row * GPT_D + tid * 4,
           (v.x - mu) * rs * wv.x + bv.x,
           (v.y - mu) * rs * wv.y + bv.y,
           (v.z - mu) * rs * wv.z + bv.z,
           (v.w - mu) * rs * wv.w + bv.w);
}

// ---------------- WMMA GEMM: C(MxN) = A(MxK) * B(KxN) (+bias, epilogue) -----
// block tile 128x128, BK=32, 8 waves (2x4), each wave 64x32 = 4x2 WMMA tiles.
// A tile staged via TDM (tensor_load_to_lds) when available.
// mode 0: Cf = acc + bias
// mode 1: Cf = Res + acc + bias          (fused residual add)
// mode 2: Cb = bf16(gelu(acc + bias))    (fused GELU for MLP)
__global__ __launch_bounds__(256) void k_gemm_bf16(
    const bf16_t* __restrict__ A, const bf16_t* __restrict__ Bw,
    const float* __restrict__ bias,
    float* __restrict__ Cf, const float* __restrict__ Res,
    bf16_t* __restrict__ Cb,
    int M, int N, int K, int mode) {
  __shared__ bf16_t sA[128][32];
  __shared__ bf16_t sB[128][32];   // transposed: [n][k] so B-frags are K-contiguous

  const int tid   = threadIdx.x;
  const int lane  = tid & 31;
  const int wave  = tid >> 5;
  const int wm    = wave >> 2;       // 0..1
  const int wn    = wave & 3;        // 0..3
  const int lrow  = lane & 15;
  const int lhalf = lane >> 4;
  const int mBase = blockIdx.y * 128;
  const int nBase = blockIdx.x * 128;

  v8f acc[4][2];
#pragma unroll
  for (int ms = 0; ms < 4; ++ms)
#pragma unroll
    for (int ns = 0; ns < 2; ++ns)
      acc[ms][ns] = (v8f){0.f, 0.f, 0.f, 0.f, 0.f, 0.f, 0.f, 0.f};

  for (int k0 = 0; k0 < K; k0 += 32) {
#if USE_TDM
    // ---- A tile 128x32 via Tensor Data Mover (wave 0 issues the DMA) ----
    if (tid < 32) {
      tdm_load_2d((unsigned)(size_t)&sA[0][0],
                  A + (size_t)mBase * K + k0,
                  (unsigned)K, 128u, 32u, 128u, (unsigned long long)K);
    }
#else
    // ---- fallback: manual A staging (b128 global -> b128 LDS) ----
#pragma unroll
    for (int i = 0; i < 2; ++i) {
      int c = tid + i * 256;             // 512 chunks of 8 bf16
      int row = c >> 2;
      int cg  = c & 3;
      uint4 v = *reinterpret_cast<const uint4*>(
          A + (size_t)(mBase + row) * K + k0 + cg * 8);
      *reinterpret_cast<uint4*>(&sA[row][cg * 8]) = v;
    }
#endif
    // ---- stage B tile 32x128 transposed into [n][k] ----
#pragma unroll
    for (int i = 0; i < 2; ++i) {
      int c = tid + i * 256;
      int kr = c >> 4;
      int cg = c & 15;
      uint4 v = *reinterpret_cast<const uint4*>(
          Bw + (size_t)(k0 + kr) * N + nBase + cg * 8);
      union { uint4 u; bf16_t h[8]; } un; un.u = v;
#pragma unroll
      for (int e = 0; e < 8; ++e) sB[cg * 8 + e][kr] = un.h[e];
    }
    if (k0 + 32 < K) {   // next-tile prefetch -> global_prefetch_b8
      __builtin_prefetch(Bw + (size_t)(k0 + 32 + (tid >> 4)) * N + nBase, 0, 0);
    }
#if USE_TDM
    if (tid < 32) __builtin_amdgcn_s_wait_tensorcnt(0);
#endif
    __syncthreads();

    v16bf af[4];
#pragma unroll
    for (int ms = 0; ms < 4; ++ms)
      af[ms] = load_afrag_row(&sA[wm * 64 + ms * 16 + lrow][0], lhalf);
    v16bf bfr[2];
#pragma unroll
    for (int ns = 0; ns < 2; ++ns)
      bfr[ns] = load_bfrag_col(&sB[wn * 32 + ns * 16 + lrow][0], lhalf);

#pragma unroll
    for (int ms = 0; ms < 4; ++ms)
#pragma unroll
      for (int ns = 0; ns < 2; ++ns)
        acc[ms][ns] = __builtin_amdgcn_wmma_f32_16x16x32_bf16(
            false, af[ms], false, bfr[ns], (short)0, acc[ms][ns], false, false);
    __syncthreads();
  }

  // ---- epilogue: C layout lane -> (M = i + 8*lhalf, N = lrow) ----
#pragma unroll
  for (int ms = 0; ms < 4; ++ms) {
#pragma unroll
    for (int ns = 0; ns < 2; ++ns) {
      int col = nBase + wn * 32 + ns * 16 + lrow;
      float bb = bias ? bias[col] : 0.0f;
#pragma unroll
      for (int i = 0; i < 8; ++i) {
        int row = mBase + wm * 64 + ms * 16 + i + 8 * lhalf;
        size_t o = (size_t)row * N + col;
        float val = acc[ms][ns][i] + bb;
        if (mode == 0) {
          Cf[o] = val;
        } else if (mode == 1) {
          Cf[o] = Res[o] + val;
        } else {
          float t = val + 0.044715f * val * val * val;
          float g = 0.5f * val * (1.0f + tanhf(0.7978845608f * t));
          Cb[o] = f2bf(g);
        }
      }
    }
  }
}

// ---------------- RoPE split: qkv f32 (B,T,3D) -> q,k,v bf16 (B,H,T,DH) -----
// Faithful to reference: theta_i = 10000 ** (-(2**i)/DH), pos = t+1.
__global__ __launch_bounds__(256) void k_rope(const float* __restrict__ qkv,
                                              bf16_t* __restrict__ qo,
                                              bf16_t* __restrict__ ko,
                                              bf16_t* __restrict__ vo) {
  int t = blockIdx.x * 256 + threadIdx.x;
  if (t >= GPT_B * GPT_T * GPT_H * 32) return;
  int j  = t & 31;
  int hh = (t >> 5) & 15;
  int tt = (t >> 9) & 1023;
  int bb = t >> 19;
  float theta = __powf(10000.0f, -exp2f((float)j) * (1.0f / 64.0f));
  float cang = (float)(tt + 1) * theta;
  float sn, cs;
  __sincosf(cang, &sn, &cs);
  size_t base = ((size_t)(bb * GPT_T + tt)) * (3 * GPT_D) + hh * GPT_DH + j;
  float q0 = qkv[base],             q1 = qkv[base + 32];
  float k0 = qkv[base + GPT_D],     k1 = qkv[base + GPT_D + 32];
  float v0 = qkv[base + 2 * GPT_D], v1 = qkv[base + 2 * GPT_D + 32];
  size_t ob = ((size_t)((bb * GPT_H + hh) * GPT_T + tt)) * GPT_DH + j;
  qo[ob] = f2bf(q0 * cs - q1 * sn); qo[ob + 32] = f2bf(q1 * cs + q0 * sn);
  ko[ob] = f2bf(k0 * cs - k1 * sn); ko[ob + 32] = f2bf(k1 * cs + k0 * sn);
  vo[ob] = f2bf(v0);                vo[ob + 32] = f2bf(v1);
}

// ---------------- wave-level causal flash attention -------------------------
// One wave (32 lanes) per (b,h, 16-query tile). Keys processed in chunks of 32.
// S = Q*K^T via 2x2 WMMA; online softmax through LDS; O += P*V via 4 WMMA.
// V chunk staged through LDS transposed so P*V B-frags are ds_load-contiguous.
__global__ __launch_bounds__(32) void k_attn(const bf16_t* __restrict__ Q,
                                             const bf16_t* __restrict__ Kk,
                                             const bf16_t* __restrict__ Vv,
                                             bf16_t* __restrict__ Y) {
  __shared__ float  sS[16][32];
  __shared__ bf16_t sP[16][32];
  __shared__ bf16_t sV[64][32];          // [dh][key_local], K-contiguous cols
  __shared__ float  sM[16], sL[16], sF[16];

  const int lane  = threadIdx.x;
  const int lrow  = lane & 15;
  const int lhalf = lane >> 4;
  const int qbase = blockIdx.x * 16;
  const int bh    = blockIdx.y;          // b*H + h
  const int b     = bh >> 4;
  const int h     = bh & 15;

  // Q tile 16x64 as two 16x32 A-fragments straight from global (row-major DH)
  const bf16_t* qp = Q + ((size_t)bh * GPT_T + qbase + lrow) * GPT_DH;
  v16bf aq[2];
#pragma unroll
  for (int c = 0; c < 2; ++c) aq[c] = load_afrag_row(qp + 32 * c, lhalf);

  v8f ov[4];
#pragma unroll
  for (int nd = 0; nd < 4; ++nd)
    ov[nd] = (v8f){0.f, 0.f, 0.f, 0.f, 0.f, 0.f, 0.f, 0.f};

  if (lane < 16) { sM[lane] = -3.0e38f; sL[lane] = 0.0f; }
  __syncthreads();

  const int nkc = (qbase + 16 + 31) >> 5;   // causal: keys <= qbase+15
  for (int c = 0; c < nkc; ++c) {
    const int kc = c * 32;
    // ---- S tiles (16 x 32), scale + causal mask, spill to LDS ----
#pragma unroll
    for (int ns = 0; ns < 2; ++ns) {
      int key  = kc + ns * 16 + lrow;
      int keyc = key < GPT_T ? key : GPT_T - 1;   // clamp OOB (masked anyway)
      const bf16_t* kp = Kk + ((size_t)bh * GPT_T + keyc) * GPT_DH;
      v8f s = (v8f){0.f, 0.f, 0.f, 0.f, 0.f, 0.f, 0.f, 0.f};
#pragma unroll
      for (int cc = 0; cc < 2; ++cc) {
        v16bf bk = load_bfrag_col(kp + 32 * cc, lhalf);
        s = __builtin_amdgcn_wmma_f32_16x16x32_bf16(
            false, aq[cc], false, bk, (short)0, s, false, false);
      }
#pragma unroll
      for (int i = 0; i < 8; ++i) {
        int qr = qbase + i + 8 * lhalf;
        float sv = s[i] * 0.125f;               // 1/sqrt(64)
        if (kc + ns * 16 + lrow > qr) sv = -3.0e38f;
        sS[i + 8 * lhalf][ns * 16 + lrow] = sv;
      }
    }
    // ---- stage V chunk transposed: sV[dh][key_local] (coalesced b128) ----
    {
      int key  = kc + lane;
      int keyc = key < GPT_T ? key : GPT_T - 1;
      const bf16_t* vp = Vv + ((size_t)bh * GPT_T + keyc) * GPT_DH;
#pragma unroll
      for (int g = 0; g < 8; ++g) {
        uint4 u = *reinterpret_cast<const uint4*>(vp + g * 8);
        union { uint4 q; bf16_t h[8]; } un; un.q = u;
#pragma unroll
        for (int e = 0; e < 8; ++e) sV[g * 8 + e][lane] = un.h[e];
      }
    }
    __syncthreads();

    // ---- online softmax per row (lanes 0..15 own a row each) ----
    if (lane < 16) {
      float mOld = sM[lane];
      float mNew = mOld;
#pragma unroll
      for (int j = 0; j < 32; ++j) mNew = fmaxf(mNew, sS[lane][j]);
      float f = __expf(mOld - mNew);
      float l = sL[lane] * f;
#pragma unroll
      for (int j = 0; j < 32; ++j) {
        float p = __expf(sS[lane][j] - mNew);
        sP[lane][j] = f2bf(p);
        l += p;
      }
      sM[lane] = mNew; sL[lane] = l; sF[lane] = f;
    }
    __syncthreads();

    // ---- rescale O, then O += P * V ----
#pragma unroll
    for (int nd = 0; nd < 4; ++nd)
#pragma unroll
      for (int i = 0; i < 8; ++i)
        ov[nd][i] *= sF[i + 8 * lhalf];

    v16bf ap = load_afrag_row(&sP[lrow][0], lhalf);
#pragma unroll
    for (int nd = 0; nd < 4; ++nd) {
      v16bf bv = load_bfrag_col(&sV[nd * 16 + lrow][0], lhalf);
      ov[nd] = __builtin_amdgcn_wmma_f32_16x16x32_bf16(
          false, ap, false, bv, (short)0, ov[nd], false, false);
    }
    __syncthreads();
  }

  // ---- normalize and write y as bf16 in (B,T,D) layout for proj GEMM ----
#pragma unroll
  for (int nd = 0; nd < 4; ++nd)
#pragma unroll
    for (int i = 0; i < 8; ++i) {
      int r  = i + 8 * lhalf;
      int qr = qbase + r;
      float val = ov[nd][i] / sL[r];
      Y[((size_t)(b * GPT_T + qr)) * GPT_D + h * GPT_DH + nd * 16 + lrow] = f2bf(val);
    }
}

// ---------------------------------------------------------------------------
extern "C" void kernel_launch(void* const* d_in, const int* in_sizes, int n_in,
                              void* d_out, int out_size, void* d_ws, size_t ws_size,
                              hipStream_t stream) {
  (void)in_sizes; (void)n_in; (void)out_size; (void)ws_size;

  const int*   idx      = (const int*)  d_in[0];
  const float* wte      = (const float*)d_in[1];
  const float* ln1_w    = (const float*)d_in[2];
  const float* ln1_b    = (const float*)d_in[3];
  const float* attn_w   = (const float*)d_in[4];
  const float* attn_b   = (const float*)d_in[5];
  const float* proj_w   = (const float*)d_in[6];
  const float* proj_b   = (const float*)d_in[7];
  const float* ln2_w    = (const float*)d_in[8];
  const float* ln2_b    = (const float*)d_in[9];
  const float* fc_w     = (const float*)d_in[10];
  const float* fc_b     = (const float*)d_in[11];
  const float* fcproj_w = (const float*)d_in[12];
  const float* fcproj_b = (const float*)d_in[13];
  const float* lnf_w    = (const float*)d_in[14];
  const float* lnf_b    = (const float*)d_in[15];
  const float* lmhead_w = (const float*)d_in[16];

  char* ws = (char*)d_ws;
  size_t off = 0;
  auto carve = [&](size_t bytes) -> char* {
    char* p = ws + off;
    off += (bytes + 255) & ~(size_t)255;
    return p;
  };

  float*  xf  = (float*) carve((size_t)GPT_M * GPT_D * 4);      // residual stream
  bf16_t* hb  = (bf16_t*)carve((size_t)GPT_M * GPT_D * 2);      // LN output
  float*  qkv = (float*) carve((size_t)GPT_M * 3 * GPT_D * 4);
  bf16_t* qb  = (bf16_t*)carve((size_t)GPT_M * GPT_D * 2);
  bf16_t* kb  = (bf16_t*)carve((size_t)GPT_M * GPT_D * 2);
  bf16_t* vb  = (bf16_t*)carve((size_t)GPT_M * GPT_D * 2);
  bf16_t* yb  = (bf16_t*)carve((size_t)GPT_M * GPT_D * 2);
  bf16_t* gb  = (bf16_t*)carve((size_t)GPT_M * 4 * GPT_D * 2);
  bf16_t* wst = (bf16_t*)carve((size_t)GPT_D * GPT_V * 2);      // weight stage

  const dim3 blk256(256);
  const dim3 blk32(32);

  // x = wte[idx]
  k_embed<<<dim3(GPT_M * GPT_D / 1024), blk256, 0, stream>>>(idx, wte, xf);

  for (int l = 0; l < GPT_L; ++l) {
    // --- attention block ---
    k_layernorm<<<dim3(GPT_M), blk256, 0, stream>>>(xf, ln1_w + l * GPT_D,
                                                    ln1_b + l * GPT_D, hb);
    {
      const int n = GPT_D * 3 * GPT_D;
      k_f32_to_bf16<<<dim3((n / 4 + 255) / 256), blk256, 0, stream>>>(
          attn_w + (size_t)l * n, wst, n / 4);
      k_gemm_bf16<<<dim3(3 * GPT_D / 128, GPT_M / 128), blk256, 0, stream>>>(
          hb, wst, attn_b + l * 3 * GPT_D, qkv, nullptr, nullptr,
          GPT_M, 3 * GPT_D, GPT_D, 0);
    }
    k_rope<<<dim3(GPT_B * GPT_T * GPT_H * 32 / 256), blk256, 0, stream>>>(
        qkv, qb, kb, vb);
    k_attn<<<dim3(GPT_T / 16, GPT_B * GPT_H), blk32, 0, stream>>>(qb, kb, vb, yb);
    {
      const int n = GPT_D * GPT_D;
      k_f32_to_bf16<<<dim3((n / 4 + 255) / 256), blk256, 0, stream>>>(
          proj_w + (size_t)l * n, wst, n / 4);
      k_gemm_bf16<<<dim3(GPT_D / 128, GPT_M / 128), blk256, 0, stream>>>(
          yb, wst, proj_b + l * GPT_D, xf, xf, nullptr,
          GPT_M, GPT_D, GPT_D, 1);                       // x += y@W + b
    }
    // --- MLP block ---
    k_layernorm<<<dim3(GPT_M), blk256, 0, stream>>>(xf, ln2_w + l * GPT_D,
                                                    ln2_b + l * GPT_D, hb);
    {
      const int n = GPT_D * 4 * GPT_D;
      k_f32_to_bf16<<<dim3((n / 4 + 255) / 256), blk256, 0, stream>>>(
          fc_w + (size_t)l * n, wst, n / 4);
      k_gemm_bf16<<<dim3(4 * GPT_D / 128, GPT_M / 128), blk256, 0, stream>>>(
          hb, wst, fc_b + l * 4 * GPT_D, nullptr, nullptr, gb,
          GPT_M, 4 * GPT_D, GPT_D, 2);                   // g = gelu(...)
      k_f32_to_bf16<<<dim3((n / 4 + 255) / 256), blk256, 0, stream>>>(
          fcproj_w + (size_t)l * n, wst, n / 4);
      k_gemm_bf16<<<dim3(GPT_D / 128, GPT_M / 128), blk256, 0, stream>>>(
          gb, wst, fcproj_b + l * GPT_D, xf, xf, nullptr,
          GPT_M, GPT_D, 4 * GPT_D, 1);                   // x += g@W + b
    }
  }

  // --- final LN + LM head ---
  k_layernorm<<<dim3(GPT_M), blk256, 0, stream>>>(xf, lnf_w, lnf_b, hb);
  {
    const size_t n = (size_t)GPT_D * GPT_V;
    k_f32_to_bf16<<<dim3((unsigned)((n / 4 + 255) / 256)), blk256, 0, stream>>>(
        lmhead_w, wst, (int)(n / 4));
    k_gemm_bf16<<<dim3(GPT_V / 128, GPT_M / 128), blk256, 0, stream>>>(
        hb, wst, nullptr, (float*)d_out, nullptr, nullptr,
        GPT_M, GPT_V, GPT_D, 0);
  }
}
